// ClearCLIPLastBlock_86912958202504
// MI455X (gfx1250) — compile-verified, matrix operand order checked by reading, not run on hardware
//
#include <hip/hip_runtime.h>
#include <hip/hip_bf16.h>
#include <stdint.h>

// ---------------------------------------------------------------------------
// ClearCLIP last block (attn 'qq' variant) for MI455X / gfx1250, wave32 WMMA.
// All GEMMs use V_WMMA_F32_16X16X32_BF16 (bf16 in, f32 accumulate).
// Attention is fused per 16-row tile so the (N,H,L,L) score tensor never
// touches HBM (682 MB avoided); scores/probabilities live in LDS.
// GEMM waves compute 32x64 tiles (8 WMMA / K-step, ~21 FLOP/B vs L2).
// Attention blocks run 4 cooperating waves -> 20 waves/WGP at 60KB LDS.
// ---------------------------------------------------------------------------

typedef __attribute__((ext_vector_type(16))) __bf16 v16bf;
typedef __attribute__((ext_vector_type(8)))  float  v8f;

union FragBF { v16bf v; uint4 q[2]; };

__device__ __forceinline__ unsigned short f2bf(float f) {
    unsigned int u = __float_as_uint(f);
    u += 0x7FFFu + ((u >> 16) & 1u);   // round-to-nearest-even
    return (unsigned short)(u >> 16);
}

static constexpr int  Cdim = 1024;
static constexpr int  Lseq = 577;
static constexpr int  Nb   = 32;
static constexpr int  Hh   = 16;
static constexpr int  Dh   = 64;
static constexpr int  LP   = 608;             // seq padded to multiple of 32
static constexpr long ROWS = (long)Nb * Lseq; // 18464 (multiple of 32)

// ---------------------------------------------------------------- zero init
__global__ void k_zero(uint4* p, long n16) {
    long i = (long)blockIdx.x * blockDim.x + threadIdx.x;
    long stride = (long)gridDim.x * blockDim.x;
    uint4 z = {0u, 0u, 0u, 0u};
    for (; i < n16; i += stride) p[i] = z;
}

// ------------------------------------------------------- fp32 -> bf16 cast
__global__ void k_f32_to_bf16(const float* __restrict__ src,
                              unsigned short* __restrict__ dst, long n4) {
    long i = (long)blockIdx.x * blockDim.x + threadIdx.x;
    long stride = (long)gridDim.x * blockDim.x;
    for (; i < n4; i += stride) {
        float4 f = ((const float4*)src)[i];
        uint2 o;
        o.x = (unsigned)f2bf(f.x) | ((unsigned)f2bf(f.y) << 16);
        o.y = (unsigned)f2bf(f.z) | ((unsigned)f2bf(f.w) << 16);
        ((uint2*)dst)[i] = o;
    }
}

// ------------------------------------------------- LayerNorm + bf16 output
__global__ __launch_bounds__(256) void k_layernorm(
        const float* __restrict__ x, const float* __restrict__ g,
        const float* __restrict__ b, unsigned short* __restrict__ z) {
    const int row = blockIdx.x;
    const int tid = threadIdx.x;
    const float4 xv = ((const float4*)(x + (long)row * Cdim))[tid];
    float s  = xv.x + xv.y + xv.z + xv.w;
    float ss = xv.x * xv.x + xv.y * xv.y + xv.z * xv.z + xv.w * xv.w;
#pragma unroll
    for (int m = 16; m >= 1; m >>= 1) {
        s  += __shfl_xor(s,  m, 32);
        ss += __shfl_xor(ss, m, 32);
    }
    __shared__ float sh1[8], sh2[8];
    const int wid = tid >> 5, lane = tid & 31;
    if (lane == 0) { sh1[wid] = s; sh2[wid] = ss; }
    __syncthreads();
    if (tid == 0) {
        float t1 = 0.f, t2 = 0.f;
        for (int i = 0; i < 8; ++i) { t1 += sh1[i]; t2 += sh2[i]; }
        sh1[0] = t1; sh2[0] = t2;
    }
    __syncthreads();
    const float mu  = sh1[0] * (1.0f / Cdim);
    const float var = sh2[0] * (1.0f / Cdim) - mu * mu;
    const float rs  = rsqrtf(var + 1e-5f);
    const float4 gv = ((const float4*)g)[tid];
    const float4 bv = ((const float4*)b)[tid];
    float z0 = (xv.x - mu) * rs * gv.x + bv.x;
    float z1 = (xv.y - mu) * rs * gv.y + bv.y;
    float z2 = (xv.z - mu) * rs * gv.z + bv.z;
    float z3 = (xv.w - mu) * rs * gv.w + bv.w;
    uint2 o;
    o.x = (unsigned)f2bf(z0) | ((unsigned)f2bf(z1) << 16);
    o.y = (unsigned)f2bf(z2) | ((unsigned)f2bf(z3) << 16);
    ((uint2*)(z + (long)row * Cdim))[tid] = o;
}

// ------------------------------------------- QKV GEMM (k-third is skipped)
// Wave computes 32 rows x 64 cols. by<16 -> q columns, by>=16 -> v columns.
// q stored (N,H,LP,D); v stored transposed (N,H,D,LP).
__global__ __launch_bounds__(32) void k_gemm_qkv(
        const unsigned short* __restrict__ Z,
        const unsigned short* __restrict__ Win,
        const float* __restrict__ bin,
        unsigned short* __restrict__ Q, unsigned short* __restrict__ VT) {
    const int lane  = threadIdx.x;
    const int l16   = lane & 15;
    const int khalf = lane >> 4;
    const int rowbase = blockIdx.x * 32;
    const int by      = blockIdx.y;
    const int wbase   = (by < 16) ? (by * 64) : (2048 + (by - 16) * 64);

    const unsigned short* arow0 = Z   + (long)(rowbase + l16) * Cdim + khalf * 8;
    const unsigned short* arow1 = arow0 + 16L * Cdim;
    const unsigned short* brow0 = Win + (long)(wbase + l16) * Cdim + khalf * 8;

    v8f acc[2][4] = {{v8f{}, v8f{}, v8f{}, v8f{}}, {v8f{}, v8f{}, v8f{}, v8f{}}};
    for (int kb = 0; kb < Cdim; kb += 32) {
        FragBF a0, a1;
        a0.q[0] = *(const uint4*)(arow0 + kb);
        a0.q[1] = *(const uint4*)(arow0 + kb + 16);
        a1.q[0] = *(const uint4*)(arow1 + kb);
        a1.q[1] = *(const uint4*)(arow1 + kb + 16);
#pragma unroll
        for (int j = 0; j < 4; ++j) {
            const unsigned short* br = brow0 + (long)j * 16 * Cdim + kb;
            FragBF b;
            b.q[0] = *(const uint4*)(br);
            b.q[1] = *(const uint4*)(br + 16);
            acc[0][j] = __builtin_amdgcn_wmma_f32_16x16x32_bf16(
                false, a0.v, false, b.v, (short)0, acc[0][j], false, false);
            acc[1][j] = __builtin_amdgcn_wmma_f32_16x16x32_bf16(
                false, a1.v, false, b.v, (short)0, acc[1][j], false, false);
        }
    }
#pragma unroll
    for (int half = 0; half < 2; ++half) {
        const int mrow0 = rowbase + half * 16 + khalf * 8;
#pragma unroll
        for (int j = 0; j < 4; ++j) {
            const int   c    = wbase + j * 16 + l16;
            const float bias = bin[c];
#pragma unroll
            for (int vg = 0; vg < 8; ++vg) {
                const int m = mrow0 + vg;
                const unsigned short bvv = f2bf(acc[half][j][vg] + bias);
                const int n = m / Lseq;
                const int l = m - n * Lseq;
                if (c < 1024) {                           // q
                    const int h = c >> 6, d = c & 63;
                    Q[(((long)(n * Hh + h) * LP) + l) * Dh + d] = bvv;
                } else {                                  // v (transposed)
                    const int cv = c - 2048;
                    const int h = cv >> 6, d = cv & 63;
                    VT[(((long)(n * Hh + h) * Dh) + d) * LP + l] = bvv;
                }
            }
        }
    }
}

// ------------------------------------- fused attention: qq^T, softmax, P*V
// 4 waves cooperate on one 16-row tile: waves split the 38 score sub-blocks,
// the 16 softmax rows, and the 4 output d-groups. ~60KB LDS per workgroup.
__global__ __launch_bounds__(128) void k_attention(
        const unsigned short* __restrict__ Q,
        const unsigned short* __restrict__ VT,
        unsigned short* __restrict__ O) {
    const int tid   = threadIdx.x;
    const int wid   = tid >> 5;
    const int lane  = tid & 31;
    const int l16   = lane & 15;
    const int khalf = lane >> 4;
    const int t = blockIdx.x, h = blockIdx.y, n = blockIdx.z;
    const int lbase = t * 16;

    __shared__ float          S[16][612];   // fp32 scores, stride dodges banks
    __shared__ unsigned short P[16][648];   // bf16 probabilities, 16B rows

    const unsigned short* Qh  = Q  + (long)(n * Hh + h) * LP * Dh;
    const unsigned short* VTh = VT + (long)(n * Hh + h) * Dh * LP;

    // A fragments: 16 q rows, K = 64 split into two 16x32 fragments
    FragBF a0, a1;
    {
        const unsigned short* qr = Qh + (long)(lbase + l16) * Dh + khalf * 8;
        a0.q[0] = *(const uint4*)(qr);       a0.q[1] = *(const uint4*)(qr + 16);
        a1.q[0] = *(const uint4*)(qr + 32);  a1.q[1] = *(const uint4*)(qr + 48);
    }

    const float scale = 0.125f;              // 1/sqrt(64)
    for (int sb = wid; sb < LP / 16; sb += 4) {  // k = q: q^T cols are q rows
        const unsigned short* kr = Qh + (long)(sb * 16 + l16) * Dh + khalf * 8;
        FragBF b0, b1;
        b0.q[0] = *(const uint4*)(kr);       b0.q[1] = *(const uint4*)(kr + 16);
        b1.q[0] = *(const uint4*)(kr + 32);  b1.q[1] = *(const uint4*)(kr + 48);
        v8f c = {};
        c = __builtin_amdgcn_wmma_f32_16x16x32_bf16(false, a0.v, false, b0.v,
                                                    (short)0, c, false, false);
        c = __builtin_amdgcn_wmma_f32_16x16x32_bf16(false, a1.v, false, b1.v,
                                                    (short)0, c, false, false);
        const int r0 = khalf * 8;
#pragma unroll
        for (int vg = 0; vg < 8; ++vg)
            S[r0 + vg][sb * 16 + l16] = c[vg] * scale;
    }
    __syncthreads();

    // softmax over valid columns [0,577); padded columns zeroed in P
    for (int r = wid; r < 16; r += 4) {
        float mx = -3.0e38f;
        for (int s = lane; s < Lseq; s += 32) mx = fmaxf(mx, S[r][s]);
#pragma unroll
        for (int m = 16; m >= 1; m >>= 1) mx = fmaxf(mx, __shfl_xor(mx, m, 32));
        float sum = 0.f;
        for (int s = lane; s < Lseq; s += 32) {
            float e = __expf(S[r][s] - mx);
            sum += e;
            S[r][s] = e;
        }
#pragma unroll
        for (int m = 16; m >= 1; m >>= 1) sum += __shfl_xor(sum, m, 32);
        const float inv = 1.0f / sum;
        for (int s = lane; s < LP; s += 32)
            P[r][s] = (s < Lseq) ? f2bf(S[r][s] * inv) : (unsigned short)0;
    }
    __syncthreads();

    // out(16 x 64) = P(16 x LP) @ V(LP x 64); wave wid owns d-group wid.
    {
        const int dg = wid;
        v8f acc = {};
        const unsigned short* vr = VTh + (long)(dg * 16 + l16) * LP + khalf * 8;
        for (int kb = 0; kb < LP; kb += 32) {
            FragBF a, b;
            const unsigned short* pr = &P[l16][kb + khalf * 8];
            a.q[0] = *(const uint4*)(pr);        a.q[1] = *(const uint4*)(pr + 16);
            b.q[0] = *(const uint4*)(vr + kb);   b.q[1] = *(const uint4*)(vr + kb + 16);
            acc = __builtin_amdgcn_wmma_f32_16x16x32_bf16(
                false, a.v, false, b.v, (short)0, acc, false, false);
        }
        const int r0 = khalf * 8;
#pragma unroll
        for (int vg = 0; vg < 8; ++vg) {
            const int l = lbase + r0 + vg;
            if (l < Lseq) {
                const int c = h * Dh + dg * 16 + l16;
                O[(long)(n * Lseq + l) * Cdim + c] = f2bf(acc[vg]);
            }
        }
    }
}

// ------------------------------------------- output projection, fp32 result
// Wave computes 32 rows x 64 cols.
__global__ __launch_bounds__(32) void k_gemm_out(
        const unsigned short* __restrict__ O,
        const unsigned short* __restrict__ Wout,
        const float* __restrict__ bout, float* __restrict__ out) {
    const int lane  = threadIdx.x;
    const int l16   = lane & 15;
    const int khalf = lane >> 4;
    const int rowbase = blockIdx.x * 32;
    const int wbase   = blockIdx.y * 64;

    const unsigned short* arow0 = O    + (long)(rowbase + l16) * Cdim + khalf * 8;
    const unsigned short* arow1 = arow0 + 16L * Cdim;
    const unsigned short* brow0 = Wout + (long)(wbase + l16) * Cdim + khalf * 8;

    v8f acc[2][4] = {{v8f{}, v8f{}, v8f{}, v8f{}}, {v8f{}, v8f{}, v8f{}, v8f{}}};
    for (int kb = 0; kb < Cdim; kb += 32) {
        FragBF a0, a1;
        a0.q[0] = *(const uint4*)(arow0 + kb);
        a0.q[1] = *(const uint4*)(arow0 + kb + 16);
        a1.q[0] = *(const uint4*)(arow1 + kb);
        a1.q[1] = *(const uint4*)(arow1 + kb + 16);
#pragma unroll
        for (int j = 0; j < 4; ++j) {
            const unsigned short* br = brow0 + (long)j * 16 * Cdim + kb;
            FragBF b;
            b.q[0] = *(const uint4*)(br);
            b.q[1] = *(const uint4*)(br + 16);
            acc[0][j] = __builtin_amdgcn_wmma_f32_16x16x32_bf16(
                false, a0.v, false, b.v, (short)0, acc[0][j], false, false);
            acc[1][j] = __builtin_amdgcn_wmma_f32_16x16x32_bf16(
                false, a1.v, false, b.v, (short)0, acc[1][j], false, false);
        }
    }
#pragma unroll
    for (int half = 0; half < 2; ++half) {
        const int mrow0 = rowbase + half * 16 + khalf * 8;
#pragma unroll
        for (int j = 0; j < 4; ++j) {
            const int   c    = wbase + j * 16 + l16;
            const float bias = bout[c];
#pragma unroll
            for (int vg = 0; vg < 8; ++vg) {
                const int m = mrow0 + vg;
                out[(long)m * Cdim + c] = acc[half][j][vg] + bias;
            }
        }
    }
}

// ---------------------------------------------------------------------------
extern "C" void kernel_launch(void* const* d_in, const int* in_sizes, int n_in,
                              void* d_out, int out_size, void* d_ws, size_t ws_size,
                              hipStream_t stream) {
    const float* x     = (const float*)d_in[0];
    const float* ln1_g = (const float*)d_in[1];
    const float* ln1_b = (const float*)d_in[2];
    const float* w_in  = (const float*)d_in[3];
    const float* b_in  = (const float*)d_in[4];
    const float* w_out = (const float*)d_in[5];
    const float* b_out = (const float*)d_in[6];
    float* out = (float*)d_out;

    char* ws = (char*)d_ws;
    // workspace layout (bytes, 256-aligned):
    unsigned short* Win_bf  = (unsigned short*)(ws);              //  6,291,456
    unsigned short* Wout_bf = (unsigned short*)(ws +   6291456);  //  2,097,152
    unsigned short* Zbf     = (unsigned short*)(ws +   8388608);  // 37,814,272
    unsigned short* Qb      = (unsigned short*)(ws +  46202880);  // 39,845,888
    unsigned short* VTb     = (unsigned short*)(ws +  86048768);  // 39,845,888
    unsigned short* Ob      = (unsigned short*)(ws + 125894656);  // 37,814,272
    // total: 163,708,928 bytes

    // zero padded q / v^T region (Qb and VTb are contiguous)
    k_zero<<<2048, 256, 0, stream>>>((uint4*)Qb, (2L * 39845888) / 16);
    // weights to bf16
    k_f32_to_bf16<<<2048, 256, 0, stream>>>(w_in,  Win_bf,  (3072L * 1024) / 4);
    k_f32_to_bf16<<<1024, 256, 0, stream>>>(w_out, Wout_bf, (1024L * 1024) / 4);
    // layernorm + cast
    k_layernorm<<<(int)ROWS, 256, 0, stream>>>(x, ln1_g, ln1_b, Zbf);
    // q/v projection (skips k), writes q (N,H,LP,D) and v^T (N,H,D,LP)
    k_gemm_qkv<<<dim3((int)(ROWS / 32), 32), 32, 0, stream>>>(Zbf, Win_bf, b_in, Qb, VTb);
    // fused attention per (16-row tile, head, batch), 4 waves cooperate
    k_attention<<<dim3(37, Hh, Nb), 128, 0, stream>>>(Qb, VTb, Ob);
    // output projection + bias, fp32
    k_gemm_out<<<dim3((int)(ROWS / 32), 16), 32, 0, stream>>>(Ob, Wout_bf, b_out, out);
}
